// AttentionBlock_57183194579607
// MI455X (gfx1250) — matrix operand order repaired and checked by reading, not compiled, not run on hardware
//
#include <hip/hip_runtime.h>

// ---------------------------------------------------------------- types
typedef __attribute__((ext_vector_type(16))) __bf16 v16bf;
typedef __attribute__((ext_vector_type(8)))  __bf16 v8bf;
typedef __attribute__((ext_vector_type(8)))  float  v8f;

// ---------------------------------------------------------------- consts
#define BB   4
#define NN   1024
#define CC   1024
#define HH   16
#define DH   64
#define M_TOK (BB * NN)            // 4096 tokens
#define K_QKV (2 * CC)             // 2048 (fused x||delta)
#define N_QKV (3 * CC)             // 3072
#define ATT_SCALE 0.125f           // 1/sqrt(64)

// GEMM macro-tile
#define MT   128                   // rows per block
#define NT   256                   // cols per block
#define PADK 40                    // LDS row stride (elements) to spread banks

// workspace layout (bytes)
#define OFF_A     ((size_t)0)                                 // bf16 [4096][2048]  16 MiB
#define OFF_W     (OFF_A  + (size_t)M_TOK * K_QKV * 2)        // bf16 [3072][2048]  12 MiB
#define OFF_QKV   (OFF_W  + (size_t)N_QKV * K_QKV * 2)        // f32  [4096][3072]  48 MiB
#define OFF_Q     (OFF_QKV+ (size_t)M_TOK * N_QKV * 4)        // bf16 [64][1024][64] 8 MiB
#define OFF_K     (OFF_Q  + (size_t)BB*HH*NN*DH * 2)
#define OFF_VT    (OFF_K  + (size_t)BB*HH*NN*DH * 2)          // bf16 [64][64][1024] (transposed V)
#define OFF_AO    (OFF_VT + (size_t)BB*HH*NN*DH * 2)          // bf16 [4096][1024]
#define OFF_WP    (OFF_AO + (size_t)M_TOK * CC * 2)           // bf16 [1024][1024]

// ---------------------------------------------------------------- helpers
static __device__ inline __bf16 f2bf(float f) {
  union { float f; unsigned u; } x; x.f = f;
  unsigned r = x.u + 0x7FFFu + ((x.u >> 16) & 1u);   // round-to-nearest-even
  unsigned short h = (unsigned short)(r >> 16);
  return __builtin_bit_cast(__bf16, h);
}

static __device__ inline v8f zero8() {
  v8f z = {0.f, 0.f, 0.f, 0.f, 0.f, 0.f, 0.f, 0.f};
  return z;
}

// A-fragment (16x32 bf16): lane l holds row (l&15),
// K = kbase..kbase+15 contiguous, kbase=(l>>4)*16.
static __device__ inline v16bf ldfragA(const __bf16* p) {
  v8bf lo = *(const v8bf*)(p);
  v8bf hi = *(const v8bf*)(p + 8);
  return __builtin_shufflevector(lo, hi, 0, 1, 2, 3, 4, 5, 6, 7,
                                 8, 9, 10, 11, 12, 13, 14, 15);
}

// B-fragment (32x16 bf16): lane l holds col (l&15), K = kbase..kbase+15
// contiguous, kbase=(l>>4)*16.  16B-aligned halves.
static __device__ inline v16bf ldfragB(const __bf16* p) {
  v8bf lo = *(const v8bf*)(p);
  v8bf hi = *(const v8bf*)(p + 8);
  return __builtin_shufflevector(lo, hi, 0, 1, 2, 3, 4, 5, 6, 7,
                                 8, 9, 10, 11, 12, 13, 14, 15);
}

static __device__ inline v8f wmma_bf16(v16bf a, v16bf b, v8f c) {
  return __builtin_amdgcn_wmma_f32_16x16x32_bf16(false, a, false, b,
                                                 (short)0, c, false, false);
}

// async global -> LDS 16B copy (ASYNCcnt path).  The LDS destination is the
// low 32 bits of the flat pointer.
static __device__ inline void async_ld_b128(__bf16* lds, const __bf16* g) {
  unsigned loff = (unsigned)(unsigned long long)(uintptr_t)lds;
  unsigned long long ga = (unsigned long long)(uintptr_t)g;
  asm volatile("global_load_async_to_lds_b128 %0, %1, off"
               :: "v"(loff), "v"(ga) : "memory");
}
#define S_WAIT_ASYNCCNT(n) asm volatile("s_wait_asynccnt " #n ::: "memory")

// ---------------------------------------------------------------- pack kernels
__global__ __launch_bounds__(256) void pack_concat_bf16(__bf16* __restrict__ dst,
                                                        const float* __restrict__ s0,
                                                        const float* __restrict__ s1) {
  size_t idx = (size_t)blockIdx.x * blockDim.x + threadIdx.x;
  size_t row = idx >> 11;
  int    k   = (int)(idx & 2047);
  float v = (k < CC) ? s0[row * CC + k] : s1[row * CC + (k - CC)];
  dst[idx] = f2bf(v);
}

__global__ __launch_bounds__(256) void pack_bf16(__bf16* __restrict__ dst,
                                                 const float* __restrict__ src) {
  size_t idx = (size_t)blockIdx.x * blockDim.x + threadIdx.x;
  dst[idx] = f2bf(src[idx]);
}

// ---------------------------------------------------------------- WMMA GEMM
// Block (8 waves) computes a 128x256 tile of C = A * Bm^T + bias.
// K stepped by 32 with double-buffered async global->LDS staging:
// steady-state loop stages the NEXT buffer (6 async b128 ops/wave), waits the
// previous one with s_wait_asynccnt 6, then runs 16 WMMAs from LDS fragments.
// The final K-step is peeled so the hot loop is branch-free.
static __device__ inline void stage_tiles(const __bf16* __restrict__ Ag,
                                          const __bf16* __restrict__ Bg,
                                          __bf16* sA, __bf16* sB,
                                          int kk, int tid, int K) {
#pragma unroll
  for (int s = 0; s < 2; ++s) {           // A: 128 rows x 4 chunks = 512
    int c = tid + s * 256;
    int row = c >> 2, col = (c & 3) * 8;
    async_ld_b128(sA + row * PADK + col, Ag + (size_t)row * K + kk + col);
  }
#pragma unroll
  for (int s = 0; s < 4; ++s) {           // B: 256 rows x 4 chunks = 1024
    int c = tid + s * 256;
    int row = c >> 2, col = (c & 3) * 8;
    async_ld_b128(sB + row * PADK + col, Bg + (size_t)row * K + kk + col);
  }
}

static __device__ inline void compute_step(const __bf16* la, const __bf16* lb,
                                           int wm, int wn, int ln, int ab, int bb,
                                           v8f acc[4][4]) {
  v16bf af[4], bfv[4];
#pragma unroll
  for (int i = 0; i < 4; ++i)
    af[i] = ldfragA(la + (wm + i * 16 + ln) * PADK + ab);
#pragma unroll
  for (int j = 0; j < 4; ++j)
    bfv[j] = ldfragB(lb + (wn + j * 16 + ln) * PADK + bb);
#pragma unroll
  for (int i = 0; i < 4; ++i)
#pragma unroll
    for (int j = 0; j < 4; ++j)
      acc[i][j] = wmma_bf16(af[i], bfv[j], acc[i][j]);
}

__global__ __launch_bounds__(256) void gemm_bf16_kernel(const __bf16* __restrict__ A,
                                                        const __bf16* __restrict__ Bm,
                                                        const float* __restrict__ bias0,
                                                        const float* __restrict__ bias1,
                                                        float* __restrict__ Cmat,
                                                        int Nout, int K, int ntilesN) {
  __shared__ __bf16 sA[2][MT * PADK];     // 20 KiB
  __shared__ __bf16 sB[2][NT * PADK];     // 40 KiB
  const int tid = threadIdx.x;
  const int w   = tid >> 5;
  const int l   = tid & 31;
  const int mt  = blockIdx.x / ntilesN;
  const int nt  = blockIdx.x % ntilesN;
  const int wm  = (w >> 2) * 64;          // wave row offset in macro-tile
  const int wn  = (w & 3) * 64;           // wave col offset in macro-tile
  const int ln  = l & 15;
  const int ab  = (l >> 4) << 4;          // A k-sub-base: 0 or 16
  const int bb  = (l >> 4) << 4;          // B k-sub-base: 0 or 16
  const int rowb = (l >> 4) << 3;         // C-frag row base

  const __bf16* Ag = A  + (size_t)(mt * MT) * K;
  const __bf16* Bg = Bm + (size_t)(nt * NT) * K;

  v8f acc[4][4];
#pragma unroll
  for (int i = 0; i < 4; ++i)
#pragma unroll
    for (int j = 0; j < 4; ++j) acc[i][j] = zero8();

  stage_tiles(Ag, Bg, sA[0], sB[0], 0, tid, K);

  int kk = 0;
  for (; kk < K - 32; kk += 32) {         // steady state: branch-free body
    const int cur = (kk >> 5) & 1;
    stage_tiles(Ag, Bg, sA[cur ^ 1], sB[cur ^ 1], kk + 32, tid, K);
    S_WAIT_ASYNCCNT(6);                   // previous buffer's 6 ops done
    __syncthreads();                      // all waves' tiles visible
    compute_step(sA[cur], sB[cur], wm, wn, ln, ab, bb, acc);
    __syncthreads();                      // reads done before next overwrite
  }
  {                                       // peeled final K-step
    const int cur = (kk >> 5) & 1;
    S_WAIT_ASYNCCNT(0);
    __syncthreads();
    compute_step(sA[cur], sB[cur], wm, wn, ln, ab, bb, acc);
  }

  const int mbase = mt * MT + wm;
  const int obase = nt * NT + wn;
#pragma unroll
  for (int j = 0; j < 4; ++j) {
    const int o = obase + j * 16 + ln;
    float bv = bias0[o];
    if (bias1) bv += bias1[o];
#pragma unroll
    for (int i = 0; i < 4; ++i) {
#pragma unroll
      for (int r = 0; r < 8; ++r) {
        const int m = mbase + i * 16 + rowb + r;
        Cmat[(size_t)m * Nout + o] = acc[i][j][r] + bv;
      }
    }
  }
}

// ---------------------------------------------------------------- LN + RoPE + V-mix
__global__ __launch_bounds__(256) void ln_rope_epilogue(
    const float* __restrict__ QKV, const float* __restrict__ rope,
    const float* __restrict__ vres,
    const float* __restrict__ qg, const float* __restrict__ qb,
    const float* __restrict__ kg, const float* __restrict__ kb,
    const float* __restrict__ lamb1, const float* __restrict__ lamb2,
    __bf16* __restrict__ Qb, __bf16* __restrict__ Kb, __bf16* __restrict__ Vt) {
  __shared__ float sBuf[256];
  const int tid  = threadIdx.x;
  const int rloc = tid >> 6;
  const int d    = tid & 63;
  const int row  = blockIdx.x * 4 + rloc;
  const int b = row >> 14, h = (row >> 10) & 15, n = row & 1023;
  const int bh = b * HH + h;

  const size_t base = ((size_t)b * NN + n) * N_QKV + h * DH + d;
  float qv = QKV[base];
  float kv = QKV[base + CC];
  float vv = QKV[base + 2 * CC];

  float vr = vres[((size_t)bh * NN + n) * DH + d];
  vv = lamb1[0] * vv + lamb2[0] * vr;
  Vt[((size_t)bh * DH + d) * NN + n] = f2bf(vv);

  const float sn = rope[(size_t)n * (2 * DH) + d];
  const float cs = rope[(size_t)n * (2 * DH) + DH + d];

  // ---- q
  sBuf[tid] = qv; __syncthreads();
  float s = 0.f, s2 = 0.f;
#pragma unroll 8
  for (int i = 0; i < 64; ++i) { float t = sBuf[rloc * 64 + i]; s += t; s2 += t * t; }
  float mean = s * (1.f / 64.f);
  float var  = s2 * (1.f / 64.f) - mean * mean;
  float qh = (qv - mean) * __frsqrt_rn(var + 1e-5f) * qg[d] + qb[d];
  __syncthreads(); sBuf[tid] = qh; __syncthreads();
  float qrot = (d < 32) ? -sBuf[rloc * 64 + d + 32] : sBuf[rloc * 64 + d - 32];
  Qb[((size_t)bh * NN + n) * DH + d] = f2bf(qh * cs + qrot * sn);
  __syncthreads();

  // ---- k
  sBuf[tid] = kv; __syncthreads();
  s = 0.f; s2 = 0.f;
#pragma unroll 8
  for (int i = 0; i < 64; ++i) { float t = sBuf[rloc * 64 + i]; s += t; s2 += t * t; }
  mean = s * (1.f / 64.f);
  var  = s2 * (1.f / 64.f) - mean * mean;
  float kh = (kv - mean) * __frsqrt_rn(var + 1e-5f) * kg[d] + kb[d];
  __syncthreads(); sBuf[tid] = kh; __syncthreads();
  float krot = (d < 32) ? -sBuf[rloc * 64 + d + 32] : sBuf[rloc * 64 + d - 32];
  Kb[((size_t)bh * NN + n) * DH + d] = f2bf(kh * cs + krot * sn);
}

// ---------------------------------------------------------------- flash attention
__global__ __launch_bounds__(32) void attn_kernel(const __bf16* __restrict__ Qb,
                                                  const __bf16* __restrict__ Kb,
                                                  const __bf16* __restrict__ Vt,
                                                  __bf16* __restrict__ attnOut) {
  __shared__ __bf16 sP[16 * 32];
  const int qt = blockIdx.x;
  const int h  = blockIdx.y;
  const int b  = blockIdx.z;
  const int bh = b * HH + h;
  const int l  = threadIdx.x;
  const int ab = (l >> 4) << 4;
  const int bb = (l >> 4) << 4;
  const int ln = l & 15;
  const int rowb = (l >> 4) << 3;

  v16bf qf[2];
#pragma unroll
  for (int f = 0; f < 2; ++f)
    qf[f] = ldfragA(Qb + ((size_t)bh * NN + qt * 16 + ln) * DH + f * 32 + ab);

  v8f oacc[4];
#pragma unroll
  for (int nj = 0; nj < 4; ++nj) oacc[nj] = zero8();
  float rmax[8], rsum[8];
#pragma unroll
  for (int r = 0; r < 8; ++r) { rmax[r] = -3.0e38f; rsum[r] = 0.f; }

  for (int kt = 0; kt < NN; kt += 32) {
    v8f s0 = zero8(), s1 = zero8();
#pragma unroll
    for (int f = 0; f < 2; ++f) {
      v16bf bk0 = ldfragB(Kb + ((size_t)bh * NN + kt + ln) * DH + f * 32 + bb);
      v16bf bk1 = ldfragB(Kb + ((size_t)bh * NN + kt + 16 + ln) * DH + f * 32 + bb);
      s0 = wmma_bf16(qf[f], bk0, s0);
      s1 = wmma_bf16(qf[f], bk1, s1);
    }

    __syncthreads();   // previous iteration's sP reads complete

#pragma unroll
    for (int r = 0; r < 8; ++r) {
      float c0 = s0[r] * ATT_SCALE;
      float c1 = s1[r] * ATT_SCALE;
      float cm = fmaxf(c0, c1);
#pragma unroll
      for (int m = 1; m < 16; m <<= 1) cm = fmaxf(cm, __shfl_xor(cm, m, 32));
      float nm = fmaxf(rmax[r], cm);
      float alpha = __expf(rmax[r] - nm);
      rmax[r] = nm;
      float p0 = __expf(c0 - nm);
      float p1 = __expf(c1 - nm);
      float ps = p0 + p1;
#pragma unroll
      for (int m = 1; m < 16; m <<= 1) ps += __shfl_xor(ps, m, 32);
      rsum[r] = rsum[r] * alpha + ps;
#pragma unroll
      for (int nj = 0; nj < 4; ++nj) oacc[nj][r] *= alpha;
      sP[(rowb + r) * 32 + ln]      = f2bf(p0);
      sP[(rowb + r) * 32 + 16 + ln] = f2bf(p1);
    }
    __syncthreads();

    v16bf pf = ldfragA(sP + ln * 32 + ab);
#pragma unroll
    for (int nj = 0; nj < 4; ++nj) {
      v16bf vf = ldfragB(Vt + ((size_t)bh * DH + nj * 16 + ln) * NN + kt + bb);
      oacc[nj] = wmma_bf16(pf, vf, oacc[nj]);
    }
  }

#pragma unroll
  for (int r = 0; r < 8; ++r) {
    float inv = 1.f / rsum[r];
    const int tok = qt * 16 + rowb + r;
#pragma unroll
    for (int nj = 0; nj < 4; ++nj) {
      const int c = h * DH + nj * 16 + ln;
      attnOut[((size_t)b * NN + tok) * CC + c] = f2bf(oacc[nj][r] * inv);
    }
  }
}

// ---------------------------------------------------------------- launcher
extern "C" void kernel_launch(void* const* d_in, const int* in_sizes, int n_in,
                              void* d_out, int out_size, void* d_ws, size_t ws_size,
                              hipStream_t stream) {
  const float* x     = (const float*)d_in[0];
  const float* rope  = (const float*)d_in[1];
  const float* dte   = (const float*)d_in[2];
  const float* vres  = (const float*)d_in[3];
  const float* Wqkv  = (const float*)d_in[4];
  const float* bqkv  = (const float*)d_in[5];
  const float* Wdt   = (const float*)d_in[6];
  const float* bdt   = (const float*)d_in[7];
  const float* qng   = (const float*)d_in[8];
  const float* qnb   = (const float*)d_in[9];
  const float* kng   = (const float*)d_in[10];
  const float* knb   = (const float*)d_in[11];
  const float* lamb1 = (const float*)d_in[12];
  const float* lamb2 = (const float*)d_in[13];
  const float* Wproj = (const float*)d_in[14];
  const float* bproj = (const float*)d_in[15];

  char* ws = (char*)d_ws;
  __bf16* Abuf   = (__bf16*)(ws + OFF_A);
  __bf16* Wbuf   = (__bf16*)(ws + OFF_W);
  float*  QKVbuf = (float*)(ws + OFF_QKV);
  __bf16* Qb     = (__bf16*)(ws + OFF_Q);
  __bf16* Kb     = (__bf16*)(ws + OFF_K);
  __bf16* Vt     = (__bf16*)(ws + OFF_VT);
  __bf16* AObuf  = (__bf16*)(ws + OFF_AO);
  __bf16* Wpb    = (__bf16*)(ws + OFF_WP);

  // 1) pack fused-K operands to bf16
  pack_concat_bf16<<<(M_TOK * K_QKV) / 256, 256, 0, stream>>>(Abuf, x, dte);
  pack_concat_bf16<<<(N_QKV * K_QKV) / 256, 256, 0, stream>>>(Wbuf, Wqkv, Wdt);
  pack_bf16<<<(CC * CC) / 256, 256, 0, stream>>>(Wpb, Wproj);

  // 2) fused QKV + delta-t GEMM: [4096,2048] x [3072,2048]^T -> f32 [4096,3072]
  gemm_bf16_kernel<<<(M_TOK / MT) * (N_QKV / NT), 256, 0, stream>>>(
      Abuf, Wbuf, bqkv, bdt, QKVbuf, N_QKV, K_QKV, N_QKV / NT);

  // 3) v-residual mix + q/k layernorm + rope
  ln_rope_epilogue<<<(BB * HH * NN) / 4, 256, 0, stream>>>(
      QKVbuf, rope, vres, qng, qnb, kng, knb, lamb1, lamb2, Qb, Kb, Vt);

  // 4) flash attention (one wave per 16-row q tile)
  attn_kernel<<<dim3(NN / 16, HH, BB), 32, 0, stream>>>(Qb, Kb, Vt, AObuf);

  // 5) output projection: [4096,1024] x [1024,1024]^T + bproj -> f32 d_out
  gemm_bf16_kernel<<<(M_TOK / MT) * (CC / NT), 256, 0, stream>>>(
      AObuf, Wpb, bproj, nullptr, (float*)d_out, CC, CC, CC / NT);
}
